// EncoderVis_841813590368
// MI455X (gfx1250) — matrix operand order repaired and checked by reading, not compiled, not run on hardware
//
#include <hip/hip_runtime.h>
#include <stdint.h>

typedef __attribute__((ext_vector_type(16))) _Float16 v16h;
typedef __attribute__((ext_vector_type(8)))  float    v8f;
typedef __attribute__((ext_vector_type(2)))  int      v2i;

#define BM 64
#define BN 64
#define KT 64
#define KP (KT + 4)      // row pitch 136B: 8B-aligned for b64, 34-word bank spread

#if defined(__has_builtin)
# if __has_builtin(__builtin_amdgcn_global_load_async_to_lds_b64)
#  define SNN_ASYNC 1
# endif
#endif
#ifndef SNN_ASYNC
# define SNN_ASYNC 0
#endif

typedef __attribute__((address_space(1))) v2i gv2i;   // global v2i32
typedef __attribute__((address_space(3))) v2i lv2i;   // LDS v2i32

union FragU { uint32_t u[8]; v16h h; };
union H2    { _Float16 h[2]; uint32_t u; };

#if SNN_ASYNC
__device__ __forceinline__ void snn_wait_async0() {
# if __has_builtin(__builtin_amdgcn_s_wait_asynccnt)
  __builtin_amdgcn_s_wait_asynccnt(0);
# else
  asm volatile("s_wait_asynccnt 0x0" ::: "memory");
# endif
}
#endif

// ---- f32 -> padded packed-f16 weight convert: out is (Ndst, Kp), zero-padded
__global__ void snn_cvt_pad(const float* __restrict__ in, uint32_t* __restrict__ out,
                            int Nsrc, int K, int Kp, long long npairs)
{
  long long i = blockIdx.x * (long long)blockDim.x + threadIdx.x;
  if (i >= npairs) return;
  int kp2 = (int)(i % (Kp / 2));
  long long n = i / (Kp / 2);
  int k = kp2 * 2;
  H2 h; h.u = 0u;
  if (n < Nsrc && k < K) {                  // K even: pair-uniform
    float2 f = *(const float2*)(in + n * (long long)K + k);
    h.h[0] = (_Float16)f.x; h.h[1] = (_Float16)f.y;
  }
  out[i] = h.u;
}

__global__ void snn_zero_u32(uint32_t* __restrict__ p, long long n)
{
  long long i = blockIdx.x * (long long)blockDim.x + threadIdx.x;
  if (i < n) p[i] = 0u;
}

// ---- x (B,F,T) f32 -> Xh (B,Tp,Fp) f16, LDS-tiled transpose, zero-padded ----
__global__ __launch_bounds__(256)
void snn_transpose_x(const float* __restrict__ X, _Float16* __restrict__ Xh,
                     int F, int T, int Fp, int Tp)
{
  __shared__ _Float16 tile[64][66];
  const int f0 = blockIdx.x * 64;
  const int t0 = blockIdx.y * 64;
  const float*  Xb  = X  + (long long)blockIdx.z * F * T;
  _Float16*     Xhb = Xh + (long long)blockIdx.z * Tp * Fp;
  const int tid = threadIdx.x;

  #pragma unroll
  for (int i = 0; i < 16; ++i) {            // read: t contiguous across lanes
    int p = tid + i * 256;
    int t = p & 63, f = p >> 6;
    int gt = t0 + t, gf = f0 + f;
    int gtc = gt < T ? gt : (T - 1);
    int gfc = gf < F ? gf : (F - 1);
    float v = Xb[(long long)gfc * T + gtc];
    tile[f][t] = (_Float16)(((gt < T) & (gf < F)) ? v : 0.0f);
  }
  __syncthreads();
  #pragma unroll
  for (int i = 0; i < 8; ++i) {             // write: f-pairs contiguous, padded
    int p = tid + i * 256;
    int fp = p & 31, t = p >> 5;
    int f = fp * 2;
    H2 h; h.h[0] = tile[f][t]; h.h[1] = tile[f + 1][t];
    *(uint32_t*)&Xhb[(long long)(t0 + t) * Fp + (f0 + f)] = h.u;
  }
}

// ---------------- WMMA GEMM, all-f16, fully padded operands ----------------
// C[t][n] = sum_k A[t][k] * W[n][k]; A rows padded to tile multiple, row
// stride ARS; W padded (rows >= n-tile span, row stride WRS = padded K).
template<int ARS, int WRS, int OST>
__global__ __launch_bounds__(256)
void snn_fc_wmma(const _Float16* __restrict__ Abase, long long aBatch,
                 const _Float16* __restrict__ W,
                 int Trows, int Nout,
                 float* __restrict__ Out, long long outBatch)
{
  __shared__ __align__(16) _Float16 Ah[2][BM][KP];
  __shared__ __align__(16) _Float16 Bh[2][BN][KP];

  const int tid  = threadIdx.x;
  const int lane = tid & 31;
  const int wv   = tid >> 5;       // 0..7
  const int wm   = wv >> 1;        // 0..3   -> 16-row subtile
  const int wnB  = (wv & 1) * 2;   // 0 or 2 -> two 16-col subtiles
  const int hi   = lane >> 4;
  const int l16  = lane & 15;

  const int m0 = blockIdx.x * BM;
  const int n0 = blockIdx.y * BN;
  const _Float16* A = Abase + (long long)blockIdx.z * aBatch;
  float*          O = Out   + (long long)blockIdx.z * outBatch;

  v8f acc0 = {}, acc1 = {};
  constexpr int ktiles = WRS / KT;          // padded K is a tile multiple

#if SNN_ASYNC
  // CDNA5 async copy: global -> LDS directly, tracked by ASYNCcnt
  auto issue = [&](int buf, int k0) {
    #pragma unroll
    for (int i = 0; i < 4; ++i) {
      int p = tid + i * 256;                // 1024 b64 chunks per tile pair
      int row = p >> 4, c = (p & 15) * 4;   // 4 halves per b64
      const _Float16* gA = A + (long long)(m0 + row) * ARS + (k0 + c);
      __builtin_amdgcn_global_load_async_to_lds_b64(
          (gv2i*)gA, (lv2i*)&Ah[buf][row][c], 0, 0);
      const _Float16* gB = W + (long long)(n0 + row) * WRS + (k0 + c);
      __builtin_amdgcn_global_load_async_to_lds_b64(
          (gv2i*)gB, (lv2i*)&Bh[buf][row][c], 0, 0);
    }
  };
#else
  uint32_t ar[8], br[8];                    // in-flight packed f16 k-pairs
  auto fetch = [&](int k0) {
    #pragma unroll
    for (int i = 0; i < 8; ++i) {
      int p = tid + i * 256;
      int k = (p & 31) * 2, m = p >> 5;
      ar[i] = *(const uint32_t*)(A + (long long)(m0 + m) * ARS + (k0 + k));
    }
    #pragma unroll
    for (int i = 0; i < 8; ++i) {
      int p = tid + i * 256;
      int k = (p & 31) * 2, n = p >> 5;
      br[i] = *(const uint32_t*)(W + (long long)(n0 + n) * WRS + (k0 + k));
    }
  };
  auto stage = [&](int buf) {
    #pragma unroll
    for (int i = 0; i < 8; ++i) {
      int p = tid + i * 256;
      int k = (p & 31) * 2, m = p >> 5;
      *(uint32_t*)&Ah[buf][m][k] = ar[i];
    }
    #pragma unroll
    for (int i = 0; i < 8; ++i) {
      int p = tid + i * 256;
      int k = (p & 31) * 2, n = p >> 5;
      *(uint32_t*)&Bh[buf][n][k] = br[i];
    }
  };
#endif

  auto compute = [&](int buf) {
    #pragma unroll
    for (int kc = 0; kc < KT / 32; ++kc) {
      FragU af, bf0, bf1;
      const uint32_t* arow = (const uint32_t*)&Ah[buf][wm * 16 + l16][kc * 32];
      #pragma unroll
      for (int j = 0; j < 8; ++j) {
        // A 16x32 f16 layout: lanes0-15 v0..3:K0-7, v4..7:K16-23; lanes16-31:+8
        int K0 = ((j >> 2) * 16) + hi * 8 + (j & 3) * 2;
        af.u[j] = arow[K0 >> 1];
      }
      const uint32_t* br0 = (const uint32_t*)&Bh[buf][(wnB + 0) * 16 + l16][kc * 32];
      const uint32_t* br1 = (const uint32_t*)&Bh[buf][(wnB + 1) * 16 + l16][kc * 32];
      #pragma unroll
      for (int j = 0; j < 8; ++j) {
        // B 32x16 f16 layout: lanes0-15 v0..3:K0-7, v4..7:K16-23; lanes16-31:+8
        int K0 = ((j >> 2) * 16) + hi * 8 + (j & 3) * 2;
        bf0.u[j] = br0[K0 >> 1];
        bf1.u[j] = br1[K0 >> 1];
      }
      acc0 = __builtin_amdgcn_wmma_f32_16x16x32_f16(false, af.h, false, bf0.h,
                                                    (short)0, acc0, false, false);
      acc1 = __builtin_amdgcn_wmma_f32_16x16x32_f16(false, af.h, false, bf1.h,
                                                    (short)0, acc1, false, false);
    }
  };

#if SNN_ASYNC
  issue(0, 0);
  snn_wait_async0();
  __syncthreads();
  for (int kt = 0; kt < ktiles; ++kt) {
    const int cur = kt & 1;
    const bool more = (kt + 1) < ktiles;
    if (more) issue(cur ^ 1, (kt + 1) * KT);   // async copies fly over compute
    compute(cur);
    if (more) {
      snn_wait_async0();
      __syncthreads();
    }
  }
#else
  fetch(0);
  stage(0);
  __syncthreads();
  for (int kt = 0; kt < ktiles; ++kt) {
    const int cur = kt & 1;
    const bool more = (kt + 1) < ktiles;
    if (more) fetch((kt + 1) * KT);
    compute(cur);
    if (more) {
      stage(cur ^ 1);
      __syncthreads();
    }
  }
#endif

  // store D: lanes 0-15 -> N=lane; VGPR r -> M = r + 8*hi
  const int gn0 = n0 + (wnB + 0) * 16 + l16;
  const int gn1 = n0 + (wnB + 1) * 16 + l16;
  #pragma unroll
  for (int r = 0; r < 8; ++r) {
    int gm = m0 + wm * 16 + r + 8 * hi;
    if (gm < Trows) {
      if (gn0 < Nout) O[(long long)gm * OST + gn0] = acc0[r];
      if (gn1 < Nout) O[(long long)gm * OST + gn1] = acc1[r];
    }
  }
}

// ---- fused scan 1: psp(u1) -> spike_dyn -> psp(s1); 2 columns/thread ----
template<int CH>
__global__ __launch_bounds__(256)
void snn_scan1(const float* __restrict__ U, _Float16* __restrict__ P2,
               int T, int BF)
{
  int id = (blockIdx.x * blockDim.x + threadIdx.x) * 2;
  if (id >= BF) return;
  const float d = 0.36787944117144233f;   // exp(-1/tau), tau = 1
  const float c = 2.718281828459045f;     // e/tau
  const float theta = 10.0f;
  const float cref = -20.0f * c;          // -scale_ref*theta * c

  float pp0 = 0.f, pq0 = 0.f, rp0 = 0.f, rq0 = 0.f, sp0 = 0.f, sq0 = 0.f;
  float pp1 = 0.f, pq1 = 0.f, rp1 = 0.f, rq1 = 0.f, sp1 = 0.f, sq1 = 0.f;
  for (int t0 = 0; t0 < T; t0 += CH) {
    float2 x[CH];
    #pragma unroll
    for (int i = 0; i < CH; ++i)
      x[i] = *(const float2*)(U + (long long)(t0 + i) * BF + id);
    uint32_t o[CH];
    #pragma unroll
    for (int i = 0; i < CH; ++i) {
      pq0 = d * (pq0 + pp0); pp0 = d * pp0 + x[i].x;
      pq1 = d * (pq1 + pp1); pp1 = d * pp1 + x[i].y;
      rq0 = d * (rq0 + rp0);
      rq1 = d * (rq1 + rp1);
      float s0 = (c * pq0 + cref * rq0 - theta >= 0.0f) ? 1.0f : 0.0f;
      float s1 = (c * pq1 + cref * rq1 - theta >= 0.0f) ? 1.0f : 0.0f;
      rp0 = d * rp0 + s0; rp1 = d * rp1 + s1;
      sq0 = d * (sq0 + sp0); sp0 = d * sp0 + s0;
      sq1 = d * (sq1 + sp1); sp1 = d * sp1 + s1;
      H2 h; h.h[0] = (_Float16)(c * sq0); h.h[1] = (_Float16)(c * sq1);
      o[i] = h.u;
    }
    #pragma unroll
    for (int i = 0; i < CH; ++i)
      *(uint32_t*)(P2 + (long long)(t0 + i) * BF + id) = o[i];
  }
}

// ---- scan 2: spike_dyn only, emit (B, O, T); per-thread output contiguous --
template<int CH>
__global__ void snn_scan2(const float* __restrict__ U2, float* __restrict__ out,
                          int T, int BO)
{
  int id = blockIdx.x * blockDim.x + threadIdx.x;
  if (id >= BO) return;
  const float d = 0.36787944117144233f;
  const float c = 2.718281828459045f;
  const float theta = 10.0f;
  const float cref = -20.0f * c;

  float rp = 0.f, rq = 0.f;
  for (int t0 = 0; t0 < T; t0 += CH) {
    float x[CH];
    #pragma unroll
    for (int i = 0; i < CH; ++i) x[i] = U2[(long long)(t0 + i) * BO + id];
    #pragma unroll
    for (int i = 0; i < CH; ++i) {
      rq = d * (rq + rp);
      float s = (x[i] + cref * rq - theta >= 0.0f) ? 1.0f : 0.0f;
      rp = d * rp + s;
      x[i] = s;
    }
    #pragma unroll
    for (int i = 0; i < CH; ++i)
      out[(long long)id * T + t0 + i] = x[i];
  }
}

extern "C" void kernel_launch(void* const* d_in, const int* in_sizes, int n_in,
                              void* d_out, int out_size, void* d_ws, size_t ws_size,
                              hipStream_t stream)
{
  const float* down = (const float*)d_in[0];  // (16, 6300, 500) == (B, F, T)
  const float* w1   = (const float*)d_in[1];  // (1024, 6300)
  const float* w2   = (const float*)d_in[2];  // (20, 1024)
  float* out = (float*)d_out;                 // (16, 20, 500)

  const int T = 500, B = 16, F = 6300, H = 1024, O = 20;
  const int Tp = 512, Fp = 6336, Op = 64;     // tile-padded dims

  // workspace layout (all chunks 256B-aligned by construction)
  char* ws = (char*)d_ws;
  size_t offU1  = 0;                                   // f32 (T,B,H)
  size_t offU2  = offU1  + (size_t)T * B * H * 4;      // f32 (T,B,O)
  size_t offXh  = offU2  + (size_t)T * B * O * 4;      // f16 (B,Tp,Fp)
  size_t offW1h = offXh  + (size_t)B * Tp * Fp * 2;    // f16 (H,Fp)
  size_t offW2h = offW1h + (size_t)H * Fp * 2;         // f16 (Op,H)
  size_t offP2h = offW2h + (size_t)Op * H * 2;         // f16 (Tp,B*H)
  size_t need   = offP2h + (size_t)Tp * B * H * 2;
  if (ws_size < need) return;
  float*     U1  = (float*)(ws + offU1);
  float*     U2  = (float*)(ws + offU2);
  _Float16*  Xh  = (_Float16*)(ws + offXh);
  _Float16*  W1h = (_Float16*)(ws + offW1h);
  _Float16*  W2h = (_Float16*)(ws + offW2h);
  _Float16*  P2h = (_Float16*)(ws + offP2h);

  // ---- per-launch f16 preparation (deterministic) ----
  long long nw1 = (long long)H * Fp / 2;
  snn_cvt_pad<<<(unsigned)((nw1 + 255) / 256), 256, 0, stream>>>(
      w1, (uint32_t*)W1h, H, F, Fp, nw1);
  long long nw2 = (long long)Op * H / 2;
  snn_cvt_pad<<<(unsigned)((nw2 + 255) / 256), 256, 0, stream>>>(
      w2, (uint32_t*)W2h, O, H, H, nw2);
  dim3 gt(Fp / 64, Tp / 64, B);
  snn_transpose_x<<<gt, 256, 0, stream>>>(down, Xh, F, T, Fp, Tp);
  long long nz = (long long)(Tp - T) * B * H / 2;      // zero P2h pad rows
  snn_zero_u32<<<(unsigned)((nz + 255) / 256), 256, 0, stream>>>(
      (uint32_t*)(P2h + (long long)T * B * H), nz);

  dim3 blk(256);

  // FC1 on raw binary spikes (psp commutes with the linear layer)
  dim3 g1(Tp / BM, H / BN, B);
  snn_fc_wmma<6336, 6336, 16384><<<g1, blk, 0, stream>>>(
      Xh, (long long)Tp * Fp, W1h, T, H, U1, (long long)H);

  // fused psp -> spike -> psp; emits packed-f16 p2
  snn_scan1<25><<<(B * H / 2 + 255) / 256, 256, 0, stream>>>(U1, P2h, T, B * H);

  // FC2 on p2: A = P2h[t][b*H + h], row stride B*H = 16384
  dim3 g2(Tp / BM, Op / BN, B);
  snn_fc_wmma<16384, 1024, 320><<<g2, blk, 0, stream>>>(
      P2h, (long long)H, W2h, T, O, U2, (long long)O);

  // final spike dynamics, transpose to (B, O, T)
  snn_scan2<50><<<1, B * O, 0, stream>>>(U2, out, T, B * O);
}
